// MultiHeadAttention_71683004170339
// MI455X (gfx1250) — matrix operand order repaired and checked by reading, not compiled, not run on hardware
//
#include <hip/hip_runtime.h>
#include <hip/hip_bf16.h>

// ---------------------------------------------------------------------------
// MHA forward for MI455X (gfx1250, wave32, WMMA).
// All matmuls use v_wmma_f32_16x16x32_bf16 (fp32 accumulate).
// attn (fp32) is materialized directly in d_out and softmaxed in place.
// ---------------------------------------------------------------------------

typedef __bf16 bf16_t;
typedef __attribute__((ext_vector_type(16))) __bf16 v16bf;
typedef __attribute__((ext_vector_type(8)))  __bf16 v8bf;
typedef __attribute__((ext_vector_type(4)))  __bf16 v4bf;
typedef __attribute__((ext_vector_type(8)))  float  v8f;

#define DMODEL 1024
#define NHEADS 16
#define HDIM   64
#define SEQ    2048
#define BATCH  2

static __device__ inline v8f wmma_bf16(v16bf a, v16bf b, v8f c) {
  // (neg_a, A, neg_b, B, c_mod, C, reuse_a, reuse_b)
  return __builtin_amdgcn_wmma_f32_16x16x32_bf16(false, a, false, b, (short)0, c,
                                                 false, false);
}

union U16bf { v16bf v; v8bf h[2]; };

// A operand (16x32 bf16, M x K): lane = row (lane&15); lanes 0-15 hold
// K=[0..7],[16..23]; lanes 16-31 hold K=[8..15],[24..31]  (ISA 7.12.2).
static __device__ inline v16bf load_a_bf16(const bf16_t* __restrict__ row,
                                           int k0, int ksel) {
  U16bf u;
  u.h[0] = *(const v8bf*)(row + k0 + 8 * ksel);
  u.h[1] = *(const v8bf*)(row + k0 + 16 + 8 * ksel);
  return u.v;
}

// B operand (32x16 bf16, K x N) sourced from an N x K row-major matrix:
// lane = column n (lane&15); lanes 0-15 hold K=[0..7],[16..23]; lanes 16-31 K=[8..15],[24..31].
static __device__ inline v16bf load_b_nk(const bf16_t* __restrict__ nrow,
                                         int k0, int ksel) {
  U16bf u;
  u.h[0] = *(const v8bf*)(nrow + k0 + 8 * ksel);
  u.h[1] = *(const v8bf*)(nrow + k0 + 16 + 8 * ksel);
  return u.v;
}

// A operand built from an fp32 row (attn matrix), converted to bf16 in-regs.
static __device__ inline v16bf load_a_f32(const float* __restrict__ row,
                                          int k0, int ksel) {
  v8f x0 = *(const v8f*)(row + k0 + 8 * ksel);
  v8f x1 = *(const v8f*)(row + k0 + 16 + 8 * ksel);
  v16bf v;
#pragma unroll
  for (int i = 0; i < 8; ++i) {
    v[i]     = (bf16_t)x0[i];
    v[i + 8] = (bf16_t)x1[i];
  }
  return v;
}

// ---------------------------------------------------------------------------
// fp32 -> bf16 elementwise convert (4 per thread)
// ---------------------------------------------------------------------------
__global__ __launch_bounds__(256) void cvt_f32_bf16(const float* __restrict__ src,
                                                    bf16_t* __restrict__ dst,
                                                    int n) {
  int i = (blockIdx.x * 256 + threadIdx.x) * 4;
  if (i < n) {
    float4 f = *(const float4*)(src + i);
    v4bf o;
    o[0] = (bf16_t)f.x; o[1] = (bf16_t)f.y; o[2] = (bf16_t)f.z; o[3] = (bf16_t)f.w;
    *(v4bf*)(dst + i) = o;
  }
}

// ---------------------------------------------------------------------------
// Projection GEMM:  C[M x N] = A[M x K] * W[N x K]^T + bias   (torch Linear)
// mode 0: Q -> bf16 [B,H,S,Dh], scaled by 1/sqrt(Dh)=0.125
// mode 1: K -> bf16 [B,H,S,Dh]
// mode 2: V -> bf16 [B,H,Dh,S]   (transposed so AV GEMM B-operand is row-major)
// mode 3: O -> fp32 [M x N]      (final output projection into d_out)
// Block = 256 thr = 8 waves (2M x 4N); wave tile 32x32 -> block tile 64x128.
// ---------------------------------------------------------------------------
__global__ __launch_bounds__(256) void proj_gemm_kernel(
    const bf16_t* __restrict__ A, const bf16_t* __restrict__ W,
    const float* __restrict__ bias, bf16_t* __restrict__ dstb,
    float* __restrict__ dstf, int K, int N, int mode) {
  const int lane = threadIdx.x & 31;
  const int wave = threadIdx.x >> 5;
  const int ksel = lane >> 4;
  const int ln16 = lane & 15;
  const int m0 = blockIdx.x * 64 + (wave >> 2) * 32;
  const int n0 = blockIdx.y * 128 + (wave & 3) * 32;

  const bf16_t* a0 = A + (size_t)(m0 + ln16) * K;
  const bf16_t* a1 = a0 + (size_t)16 * K;
  const bf16_t* b0 = W + (size_t)(n0 + ln16) * K;
  const bf16_t* b1 = b0 + (size_t)16 * K;

  v8f acc[2][2] = {};
  for (int k0 = 0; k0 < K; k0 += 32) {
    __builtin_prefetch(a0 + k0 + 64, 0, 0);   // global_prefetch_b8
    v16bf va0 = load_a_bf16(a0, k0, ksel);
    v16bf va1 = load_a_bf16(a1, k0, ksel);
    v16bf vb0 = load_b_nk(b0, k0, ksel);
    v16bf vb1 = load_b_nk(b1, k0, ksel);
    acc[0][0] = wmma_bf16(va0, vb0, acc[0][0]);
    acc[0][1] = wmma_bf16(va0, vb1, acc[0][1]);
    acc[1][0] = wmma_bf16(va1, vb0, acc[1][0]);
    acc[1][1] = wmma_bf16(va1, vb1, acc[1][1]);
  }

#pragma unroll
  for (int mi = 0; mi < 2; ++mi) {
#pragma unroll
    for (int ni = 0; ni < 2; ++ni) {
      const int col = n0 + ni * 16 + ln16;
      const float bv = bias[col];
#pragma unroll
      for (int i = 0; i < 8; ++i) {
        const int row = m0 + mi * 16 + ksel * 8 + i;  // D layout: lanes16-31 = M+8
        float val = acc[mi][ni][i] + bv;
        if (mode == 3) {
          dstf[(size_t)row * N + col] = val;
        } else {
          const int b = row / SEQ, s = row % SEQ;
          const int h = col >> 6, d = col & 63;
          if (mode == 0)
            dstb[((size_t)(b * NHEADS + h) * SEQ + s) * HDIM + d] =
                (bf16_t)(val * 0.125f);
          else if (mode == 1)
            dstb[((size_t)(b * NHEADS + h) * SEQ + s) * HDIM + d] = (bf16_t)val;
          else
            dstb[((size_t)(b * NHEADS + h) * HDIM + d) * SEQ + s] = (bf16_t)val;
        }
      }
    }
  }
}

// ---------------------------------------------------------------------------
// scores[bh, si, sj] = sum_d Qs[bh,si,d] * K[bh,sj,d]   (Q pre-scaled)
// M=N=SEQ, K=HDIM. Writes fp32 straight into d_out attn region.
// ---------------------------------------------------------------------------
__global__ __launch_bounds__(256) void attn_scores_kernel(
    const bf16_t* __restrict__ Qh, const bf16_t* __restrict__ Kh,
    float* __restrict__ out) {
  const int lane = threadIdx.x & 31;
  const int wave = threadIdx.x >> 5;
  const int ksel = lane >> 4;
  const int ln16 = lane & 15;
  const int bh = blockIdx.z;
  const int m0 = blockIdx.x * 64 + (wave >> 2) * 32;
  const int n0 = blockIdx.y * 128 + (wave & 3) * 32;

  const bf16_t* Q  = Qh + (size_t)bh * SEQ * HDIM;
  const bf16_t* Kp = Kh + (size_t)bh * SEQ * HDIM;
  float* O = out + (size_t)bh * SEQ * SEQ;

  const bf16_t* a0 = Q + (size_t)(m0 + ln16) * HDIM;
  const bf16_t* a1 = a0 + 16 * HDIM;
  const bf16_t* b0 = Kp + (size_t)(n0 + ln16) * HDIM;
  const bf16_t* b1 = b0 + 16 * HDIM;

  v8f acc[2][2] = {};
#pragma unroll
  for (int k0 = 0; k0 < HDIM; k0 += 32) {
    v16bf va0 = load_a_bf16(a0, k0, ksel);
    v16bf va1 = load_a_bf16(a1, k0, ksel);
    v16bf vb0 = load_b_nk(b0, k0, ksel);
    v16bf vb1 = load_b_nk(b1, k0, ksel);
    acc[0][0] = wmma_bf16(va0, vb0, acc[0][0]);
    acc[0][1] = wmma_bf16(va0, vb1, acc[0][1]);
    acc[1][0] = wmma_bf16(va1, vb0, acc[1][0]);
    acc[1][1] = wmma_bf16(va1, vb1, acc[1][1]);
  }

#pragma unroll
  for (int mi = 0; mi < 2; ++mi)
#pragma unroll
    for (int ni = 0; ni < 2; ++ni)
#pragma unroll
      for (int i = 0; i < 8; ++i)
        O[(size_t)(m0 + mi * 16 + ksel * 8 + i) * SEQ + (n0 + ni * 16 + ln16)] =
            acc[mi][ni][i];
}

// ---------------------------------------------------------------------------
// In-place row softmax over SEQ=2048 fp32 (one block of 256 per row).
// ---------------------------------------------------------------------------
__global__ __launch_bounds__(256) void softmax_kernel(float* __restrict__ attn) {
  float* p = attn + (size_t)blockIdx.x * SEQ;
  const int t = threadIdx.x;
  float r[8];
#pragma unroll
  for (int i = 0; i < 8; ++i) r[i] = p[t + 256 * i];

  __shared__ float red[256];
  float m = r[0];
#pragma unroll
  for (int i = 1; i < 8; ++i) m = fmaxf(m, r[i]);
  red[t] = m;
  __syncthreads();
  for (int off = 128; off > 0; off >>= 1) {
    if (t < off) red[t] = fmaxf(red[t], red[t + off]);
    __syncthreads();
  }
  m = red[0];
  __syncthreads();

  float s = 0.f;
#pragma unroll
  for (int i = 0; i < 8; ++i) { r[i] = __expf(r[i] - m); s += r[i]; }
  red[t] = s;
  __syncthreads();
  for (int off = 128; off > 0; off >>= 1) {
    if (t < off) red[t] += red[t + off];
    __syncthreads();
  }
  const float inv = 1.0f / red[0];
#pragma unroll
  for (int i = 0; i < 8; ++i) p[t + 256 * i] = r[i] * inv;
}

// ---------------------------------------------------------------------------
// out_head[bh,s,d] = sum_k attn[bh,s,k] * Vt[bh,d,k]
// A = attn fp32 (converted to bf16 in regs), B = Vt bf16 row-major (d-rows).
// Wave tile 32(M) x 64(N=full Dh); 8 waves per block -> 256 rows per block.
// Output written as bf16 [B,S,DMODEL] for the final projection.
// ---------------------------------------------------------------------------
__global__ __launch_bounds__(256) void attn_v_kernel(
    const float* __restrict__ attn, const bf16_t* __restrict__ Vt,
    bf16_t* __restrict__ Obf) {
  const int lane = threadIdx.x & 31;
  const int wave = threadIdx.x >> 5;
  const int ksel = lane >> 4;
  const int ln16 = lane & 15;
  const int bh = blockIdx.z;
  const int b = bh >> 4, h = bh & 15;
  const int m0 = blockIdx.x * 256 + wave * 32;

  const float*  Ab = attn + (size_t)bh * SEQ * SEQ;
  const bf16_t* Vb = Vt + (size_t)bh * HDIM * SEQ;

  const float* a0 = Ab + (size_t)(m0 + ln16) * SEQ;
  const float* a1 = a0 + (size_t)16 * SEQ;
  const bf16_t* br[4];
#pragma unroll
  for (int ni = 0; ni < 4; ++ni) br[ni] = Vb + (size_t)(ni * 16 + ln16) * SEQ;

  v8f acc[2][4] = {};
  for (int k0 = 0; k0 < SEQ; k0 += 32) {
    __builtin_prefetch(a0 + k0 + 64, 0, 0);   // global_prefetch_b8
    v16bf va0 = load_a_f32(a0, k0, ksel);
    v16bf va1 = load_a_f32(a1, k0, ksel);
#pragma unroll
    for (int ni = 0; ni < 4; ++ni) {
      v16bf vb = load_b_nk(br[ni], k0, ksel);
      acc[0][ni] = wmma_bf16(va0, vb, acc[0][ni]);
      acc[1][ni] = wmma_bf16(va1, vb, acc[1][ni]);
    }
  }

#pragma unroll
  for (int mi = 0; mi < 2; ++mi)
#pragma unroll
    for (int ni = 0; ni < 4; ++ni) {
      const int d = ni * 16 + ln16;
#pragma unroll
      for (int i = 0; i < 8; ++i) {
        const int s = m0 + mi * 16 + ksel * 8 + i;
        Obf[(size_t)(b * SEQ + s) * DMODEL + h * HDIM + d] = (bf16_t)acc[mi][ni][i];
      }
    }
}

// ---------------------------------------------------------------------------
extern "C" void kernel_launch(void* const* d_in, const int* in_sizes, int n_in,
                              void* d_out, int out_size, void* d_ws,
                              size_t ws_size, hipStream_t stream) {
  (void)in_sizes; (void)n_in; (void)out_size; (void)ws_size;

  const float* query = (const float*)d_in[0];
  const float* key_  = (const float*)d_in[1];
  const float* value = (const float*)d_in[2];
  const float* Wq = (const float*)d_in[3];
  const float* bq = (const float*)d_in[4];
  const float* Wk = (const float*)d_in[5];
  const float* bk = (const float*)d_in[6];
  const float* Wv = (const float*)d_in[7];
  const float* bv = (const float*)d_in[8];
  const float* Wo = (const float*)d_in[9];
  const float* bo = (const float*)d_in[10];

  const int MROWS = BATCH * SEQ;               // 4096
  const int NX = MROWS * DMODEL;               // 4 Mi elems
  const int NW = DMODEL * DMODEL;              // 1 Mi elems
  const size_t MB = 1u << 20;

  char* ws = (char*)d_ws;
  bf16_t* Xq  = (bf16_t*)(ws + 0 * MB);
  bf16_t* Xk  = (bf16_t*)(ws + 8 * MB);
  bf16_t* Xv  = (bf16_t*)(ws + 16 * MB);
  bf16_t* Wqb = (bf16_t*)(ws + 24 * MB);
  bf16_t* Wkb = (bf16_t*)(ws + 26 * MB);
  bf16_t* Wvb = (bf16_t*)(ws + 28 * MB);
  bf16_t* Wob = (bf16_t*)(ws + 30 * MB);
  bf16_t* Qh  = (bf16_t*)(ws + 32 * MB);  // [B,H,S,Dh], scaled
  bf16_t* Kh  = (bf16_t*)(ws + 40 * MB);  // [B,H,S,Dh]
  bf16_t* Vt  = (bf16_t*)(ws + 48 * MB);  // [B,H,Dh,S]
  bf16_t* Obf = (bf16_t*)(ws + 56 * MB);  // [B*S, DMODEL]

  // fp32 -> bf16 staging
  cvt_f32_bf16<<<NX / 1024, 256, 0, stream>>>(query, Xq, NX);
  cvt_f32_bf16<<<NX / 1024, 256, 0, stream>>>(key_,  Xk, NX);
  cvt_f32_bf16<<<NX / 1024, 256, 0, stream>>>(value, Xv, NX);
  cvt_f32_bf16<<<NW / 1024, 256, 0, stream>>>(Wq, Wqb, NW);
  cvt_f32_bf16<<<NW / 1024, 256, 0, stream>>>(Wk, Wkb, NW);
  cvt_f32_bf16<<<NW / 1024, 256, 0, stream>>>(Wv, Wvb, NW);
  cvt_f32_bf16<<<NW / 1024, 256, 0, stream>>>(Wo, Wob, NW);

  // Q/K/V projections (WMMA)
  dim3 gproj(MROWS / 64, DMODEL / 128);
  proj_gemm_kernel<<<gproj, 256, 0, stream>>>(Xq, Wqb, bq, Qh, nullptr,
                                              DMODEL, DMODEL, 0);
  proj_gemm_kernel<<<gproj, 256, 0, stream>>>(Xk, Wkb, bk, Kh, nullptr,
                                              DMODEL, DMODEL, 1);
  proj_gemm_kernel<<<gproj, 256, 0, stream>>>(Xv, Wvb, bv, Vt, nullptr,
                                              DMODEL, DMODEL, 2);

  float* out  = (float*)d_out;
  float* attn = out + (size_t)MROWS * DMODEL;  // attn region of d_out

  // scores = (Q/sqrt(d)) K^T  -> d_out attn region (WMMA)
  dim3 gsc(SEQ / 64, SEQ / 128, BATCH * NHEADS);
  attn_scores_kernel<<<gsc, 256, 0, stream>>>(Qh, Kh, attn);

  // in-place softmax
  softmax_kernel<<<BATCH * NHEADS * SEQ, 256, 0, stream>>>(attn);

  // attn @ V (WMMA, attn converted to bf16 in registers)
  dim3 gav(SEQ / 256, 1, BATCH * NHEADS);
  attn_v_kernel<<<gav, 256, 0, stream>>>(attn, Vt, Obf);

  // output projection -> d_out[0 .. B*S*D)
  proj_gemm_kernel<<<gproj, 256, 0, stream>>>(Obf, Wob, bo, nullptr, out,
                                              DMODEL, DMODEL, 3);
}
